// CGCNN_15573551415580
// MI455X (gfx1250) — matrix operand order repaired and checked by reading, not compiled, not run on hardware
//
#include <hip/hip_runtime.h>
#include <hip/hip_bf16.h>
#include <stdint.h>

// ---------------------------------------------------------------------------
// CGCNN forward on gfx1250 (MI455X). Edge GEMMs via V_WMMA_F32_16X16X32_F16.
// Each wave: 32-edge tile (two 16x32 A fragments) so every B-fragment LDS
// load feeds two WMMAs (halves dscnt stalls and LDS B traffic).
// ---------------------------------------------------------------------------

typedef __attribute__((ext_vector_type(16))) _Float16 v16h;
typedef __attribute__((ext_vector_type(8)))  float    v8f;

#define HID   64
#define ZDIM  169
#define KPAD  192           // 6 K-tiles of 32
#define NKT   6
#define NNT   4             // 64 cols = 4 tiles of 16
#define FRAG  512           // halves per fragment (32 lanes * 16)
#define WPERL 24576         // halves per layer: 2 mats * 6 kt * 4 nt * 512
#define WAVES 8             // waves per block in conv kernel
#define ZWAVE (2 * NKT * FRAG)   // halves of z-staging per wave (two M-tiles)

// ---------------------------------------------------------------------------
// Build f16 B-matrix fragments for W_sig / W_sp.
// Layout: [layer][mat][kt][nt][lane][j]   (j = 16 contiguous halves per lane)
// B element (k_local, n):  lane = (k_local>=16)*16 + n ; j = k_local & 15
// ---------------------------------------------------------------------------
__global__ void build_wfrag(const float* __restrict__ W_sig,
                            const float* __restrict__ W_sp,
                            _Float16* __restrict__ wfrag, int total)
{
    int idx = blockIdx.x * blockDim.x + threadIdx.x;
    if (idx >= total) return;
    int t   = idx;
    int j   = t & 15;  t >>= 4;
    int ln  = t & 31;  t >>= 5;
    int nt  = t & 3;   t >>= 2;
    int kt  = t % NKT; t /= NKT;
    int mat = t & 1;
    int layer = t >> 1;

    int n  = ln & 15;
    int kl = ((ln >> 4) << 4) + j;   // k within 32-tile
    int k  = kt * 32 + kl;
    int col = nt * 16 + n;
    float v = 0.0f;
    if (k < ZDIM) {
        const float* W = mat ? W_sp : W_sig;
        v = W[((size_t)layer * ZDIM + k) * HID + col];
    }
    wfrag[idx] = (_Float16)v;
}

// ---------------------------------------------------------------------------
// h = x @ W_embed + b_embed      [N,92] @ [92,64]
// ---------------------------------------------------------------------------
__global__ void embed_kernel(const float* __restrict__ x,
                             const float* __restrict__ W,
                             const float* __restrict__ b,
                             float* __restrict__ h, int N)
{
    size_t total  = (size_t)N * HID;
    size_t stride = (size_t)gridDim.x * blockDim.x;
    for (size_t i = blockIdx.x * (size_t)blockDim.x + threadIdx.x; i < total; i += stride) {
        int node = (int)(i >> 6);
        int col  = (int)(i & 63);
        float acc = b[col];
        const float* xr = x + (size_t)node * 92;
        #pragma unroll 4
        for (int k = 0; k < 92; ++k)
            acc = fmaf(xr[k], W[(size_t)k * HID + col], acc);
        h[i] = acc;
    }
}

__global__ void zero_kernel(float* __restrict__ p, size_t n)
{
    size_t stride = (size_t)gridDim.x * blockDim.x;
    for (size_t i = blockIdx.x * (size_t)blockDim.x + threadIdx.x; i < n; i += stride)
        p[i] = 0.0f;
}

// ---------------------------------------------------------------------------
// Conv layer edge kernel.  One wave per 32-edge tile (two 16-row A tiles).
//   z = [h[src] | h[dst] | edge_attr | pad]  (169 -> 192, f16)
//   gated = sigmoid(z@Wsig+b), msg = softplus(z@Wsp+b)
//   atomicAdd(agg[dst], gated*msg)
// ---------------------------------------------------------------------------
__global__ __launch_bounds__(256)
void conv_edges(const float* __restrict__ h,
                const int* __restrict__ src,
                const int* __restrict__ dst,
                const float* __restrict__ edge_attr,
                const _Float16* __restrict__ wfrag,   // this layer's fragments
                const float* __restrict__ bsig,
                const float* __restrict__ bsp,
                float* __restrict__ agg, int E)
{
    __shared__ __align__(64) _Float16 smemZ[WAVES * ZWAVE];  // 96 KB
    __shared__ __align__(64) _Float16 smemW[WPERL];          // 48 KB
    __shared__ int sdst[WAVES][32];

    const int tid  = threadIdx.x;
    const int lane = tid & 31;
    const int wid  = tid >> 5;

    // cooperative weight load -> LDS (48 KB, uint4 = 16B each)
    {
        const uint4* wsrc = (const uint4*)wfrag;
        uint4* wdstp = (uint4*)smemW;
        for (int i = tid; i < WPERL * 2 / 16; i += 256)
            wdstp[i] = wsrc[i];
    }
    __syncthreads();

    const int ntiles  = (E + 31) >> 5;
    const int wglobal = blockIdx.x * WAVES + wid;
    const int wstride = gridDim.x * WAVES;

    // gather: one lane per edge; lane>=16 fills the second M-tile
    const int m_sub = lane & 15;          // row within A tile
    const int tsel  = lane >> 4;          // which of the two A tiles
    _Float16* zbase = &smemZ[wid * ZWAVE + tsel * (NKT * FRAG)];

    for (int tile = wglobal; tile < ntiles; tile += wstride) {
        // ------------------- gather z tile into swizzled A layout ----------
        size_t e = (size_t)tile * 32 + lane;
        bool valid = (e < (size_t)E);
        int s = 0, d = 0;
        if (valid) { s = src[e]; d = dst[e]; }
        sdst[wid][lane] = valid ? d : -1;

        // (m_sub, c) -> LDS half index (A fragment layout, 16x32 f16, wave32)
        auto stz = [&](int c, float v) {
            int kt = c >> 5, kl = c & 31;
            int tl = (((kl >> 4) & 1) << 4) + m_sub;             // lane
            int j  = kl & 15;                                    // half idx
            zbase[kt * FRAG + tl * 16 + j] = (_Float16)v;
        };
        const float4 z4 = make_float4(0.f, 0.f, 0.f, 0.f);
        {
            const float4* hs = (const float4*)(h + (size_t)s * HID);
            #pragma unroll
            for (int q = 0; q < 16; ++q) {
                float4 v = valid ? hs[q] : z4;
                stz(4*q+0, v.x); stz(4*q+1, v.y); stz(4*q+2, v.z); stz(4*q+3, v.w);
            }
            const float4* hd = (const float4*)(h + (size_t)d * HID);
            #pragma unroll
            for (int q = 0; q < 16; ++q) {
                float4 v = valid ? hd[q] : z4;
                int c = 64 + 4*q;
                stz(c+0, v.x); stz(c+1, v.y); stz(c+2, v.z); stz(c+3, v.w);
            }
            const float* ea = edge_attr + e * 41;
            #pragma unroll
            for (int c = 128; c < 128 + 41; ++c)
                stz(c, valid ? ea[c - 128] : 0.0f);
            #pragma unroll
            for (int c = 169; c < KPAD; ++c) stz(c, 0.0f);
        }

        // ------------------- WMMA: 6 kt x 4 nt x 2 mats x 2 M-tiles --------
        v8f accS[2][NNT], accP[2][NNT];
        #pragma unroll
        for (int mt = 0; mt < 2; ++mt)
            #pragma unroll
            for (int nt = 0; nt < NNT; ++nt) {
                accS[mt][nt] = (v8f)0.0f;
                accP[mt][nt] = (v8f)0.0f;
            }

        const _Float16* zw = &smemZ[wid * ZWAVE];
        #pragma unroll
        for (int kt = 0; kt < NKT; ++kt) {
            v16h a0 = *(const v16h*)&zw[kt * FRAG + lane * 16];
            v16h a1 = *(const v16h*)&zw[(NKT + kt) * FRAG + lane * 16];
            #pragma unroll
            for (int nt = 0; nt < NNT; ++nt) {
                v16h bS = *(const v16h*)&smemW[((0 * NKT + kt) * NNT + nt) * FRAG + lane * 16];
                accS[0][nt] = __builtin_amdgcn_wmma_f32_16x16x32_f16(
                    false, a0, false, bS, (short)0, accS[0][nt], false, false);
                accS[1][nt] = __builtin_amdgcn_wmma_f32_16x16x32_f16(
                    false, a1, false, bS, (short)0, accS[1][nt], false, false);
                v16h bP = *(const v16h*)&smemW[((1 * NKT + kt) * NNT + nt) * FRAG + lane * 16];
                accP[0][nt] = __builtin_amdgcn_wmma_f32_16x16x32_f16(
                    false, a0, false, bP, (short)0, accP[0][nt], false, false);
                accP[1][nt] = __builtin_amdgcn_wmma_f32_16x16x32_f16(
                    false, a1, false, bP, (short)0, accP[1][nt], false, false);
            }
        }

        // ------------------- epilogue: bias, act, scatter-add --------------
        // C/D layout: VGPR v -> row m = v + 8*(lane>=16) ; col = lane&15
        const int nlo  = lane & 15;
        const int mofs = (lane >> 4) << 3;
        #pragma unroll
        for (int mt = 0; mt < 2; ++mt) {
            #pragma unroll
            for (int nt = 0; nt < NNT; ++nt) {
                int col = nt * 16 + nlo;
                float bS = bsig[col], bP = bsp[col];
                #pragma unroll
                for (int v = 0; v < 8; ++v) {
                    int dnode = sdst[wid][mt * 16 + mofs + v];
                    if (dnode >= 0) {
                        float g  = 1.0f / (1.0f + __expf(-(accS[mt][nt][v] + bS)));
                        float xm = accP[mt][nt][v] + bP;
                        float sp = fmaxf(xm, 0.0f) + log1pf(__expf(-fabsf(xm)));
                        atomicAdd(&agg[(size_t)dnode * HID + col], g * sp);
                    }
                }
            }
        }
    }
}

// ---------------------------------------------------------------------------
// h += agg; accumulate per-column sum / sumsq into global stats.
// Thread's column is fixed (tid & 63) since stride is a multiple of 64.
// ---------------------------------------------------------------------------
__global__ __launch_bounds__(256)
void residual_stats(float* __restrict__ h, const float* __restrict__ agg,
                    float* __restrict__ gsum, float* __restrict__ gsumsq, int N)
{
    const int tid = threadIdx.x;
    float s = 0.f, q = 0.f;
    size_t total  = (size_t)N * HID;
    size_t stride = (size_t)gridDim.x * 256;
    for (size_t i = blockIdx.x * (size_t)256 + tid; i < total; i += stride) {
        float v = h[i] + agg[i];
        h[i] = v;
        s += v;
        q = fmaf(v, v, q);
    }
    __shared__ float ls[256], lq[256];
    ls[tid] = s; lq[tid] = q;
    __syncthreads();
    if (tid < 64) {
        float S = ls[tid] + ls[tid + 64] + ls[tid + 128] + ls[tid + 192];
        float Q = lq[tid] + lq[tid + 64] + lq[tid + 128] + lq[tid + 192];
        atomicAdd(&gsum[tid], S);
        atomicAdd(&gsumsq[tid], Q);
    }
}

__global__ void bn_apply(float* __restrict__ h,
                         const float* __restrict__ gsum,
                         const float* __restrict__ gsumsq,
                         const float* __restrict__ gamma,
                         const float* __restrict__ beta, int N)
{
    size_t total  = (size_t)N * HID;
    size_t stride = (size_t)gridDim.x * blockDim.x;
    float invN = 1.0f / (float)N;
    for (size_t i = blockIdx.x * (size_t)blockDim.x + threadIdx.x; i < total; i += stride) {
        int col = (int)(i & 63);
        float mean = gsum[col] * invN;
        float var  = gsumsq[col] * invN - mean * mean;
        h[i] = (h[i] - mean) * rsqrtf(var + 1e-5f) * gamma[col] + beta[col];
    }
}

__global__ void pool_kernel(const float* __restrict__ h,
                            const int* __restrict__ graph_idx,
                            float* __restrict__ pooled, int N)
{
    size_t total  = (size_t)N * HID;
    size_t stride = (size_t)gridDim.x * blockDim.x;
    for (size_t i = blockIdx.x * (size_t)blockDim.x + threadIdx.x; i < total; i += stride) {
        int node = (int)(i >> 6);
        int col  = (int)(i & 63);
        atomicAdd(&pooled[(size_t)graph_idx[node] * HID + col], h[i]);
    }
}

// out[g] = (pooled[g] @ W_fc + b_fc) @ W_out + b_out     (128 threads / graph)
__global__ __launch_bounds__(128)
void readout_kernel(const float* __restrict__ pooled,
                    const float* __restrict__ W_fc, const float* __restrict__ b_fc,
                    const float* __restrict__ W_out, const float* __restrict__ b_out,
                    float* __restrict__ out)
{
    const int g = blockIdx.x;
    const int p = threadIdx.x;
    const float* pg = pooled + (size_t)g * HID;
    float acc = b_fc[p];
    #pragma unroll
    for (int k = 0; k < HID; ++k)
        acc = fmaf(pg[k], W_fc[(size_t)k * 128 + p], acc);
    float val = acc * W_out[p];
    __shared__ float red[128];
    red[p] = val;
    __syncthreads();
    for (int s = 64; s > 0; s >>= 1) {
        if (p < s) red[p] += red[p + s];
        __syncthreads();
    }
    if (p == 0) out[g] = red[0] + b_out[0];
}

// ---------------------------------------------------------------------------
extern "C" void kernel_launch(void* const* d_in, const int* in_sizes, int n_in,
                              void* d_out, int out_size, void* d_ws, size_t ws_size,
                              hipStream_t stream)
{
    const float* x         = (const float*)d_in[0];
    const float* edge_attr = (const float*)d_in[1];
    const int*   src       = (const int*)d_in[2];
    const int*   dst       = (const int*)d_in[3];
    const int*   graph_idx = (const int*)d_in[4];
    const float* W_embed   = (const float*)d_in[6];
    const float* b_embed   = (const float*)d_in[7];
    const float* W_sig     = (const float*)d_in[8];
    const float* b_sig     = (const float*)d_in[9];
    const float* W_sp      = (const float*)d_in[10];
    const float* b_sp      = (const float*)d_in[11];
    const float* bn_gamma  = (const float*)d_in[12];
    const float* bn_beta   = (const float*)d_in[13];
    const float* W_fc      = (const float*)d_in[14];
    const float* b_fc      = (const float*)d_in[15];
    const float* W_out     = (const float*)d_in[16];
    const float* b_out     = (const float*)d_in[17];

    const int N = in_sizes[0] / 92;
    const int E = in_sizes[2];
    const int G = out_size;            // NOUT == 1
    const int NCONV = 3;

    // workspace partition (256B aligned)
    uintptr_t p = (uintptr_t)d_ws;
    auto alloc = [&](size_t bytes) {
        p = (p + 255) & ~(uintptr_t)255;
        void* r = (void*)p;
        p += bytes;
        return r;
    };
    float*     h      = (float*)alloc((size_t)N * HID * sizeof(float));
    float*     agg    = (float*)alloc((size_t)N * HID * sizeof(float));
    float*     pooled = (float*)alloc((size_t)G * HID * sizeof(float));
    float*     stats  = (float*)alloc(128 * sizeof(float));          // sum[64], sumsq[64]
    _Float16*  wfrag  = (_Float16*)alloc((size_t)NCONV * WPERL * sizeof(_Float16));

    // 1) weight fragments (f32 -> f16 B layout), once per launch
    {
        int total = NCONV * WPERL;
        build_wfrag<<<(total + 255) / 256, 256, 0, stream>>>(W_sig, W_sp, wfrag, total);
    }
    // 2) node embedding
    embed_kernel<<<2048, 256, 0, stream>>>(x, W_embed, b_embed, h, N);

    // 3) conv layers
    for (int i = 0; i < NCONV; ++i) {
        zero_kernel<<<2048, 256, 0, stream>>>(agg, (size_t)N * HID);
        zero_kernel<<<1, 128, 0, stream>>>(stats, 128);
        conv_edges<<<512, 256, 0, stream>>>(h, src, dst, edge_attr,
                                            wfrag + (size_t)i * WPERL,
                                            b_sig + (size_t)i * HID,
                                            b_sp  + (size_t)i * HID,
                                            agg, E);
        residual_stats<<<512, 256, 0, stream>>>(h, agg, stats, stats + 64, N);
        bn_apply<<<2048, 256, 0, stream>>>(h, stats, stats + 64,
                                           bn_gamma + (size_t)i * HID,
                                           bn_beta  + (size_t)i * HID, N);
    }

    // 4) graph pooling + readout
    zero_kernel<<<64, 256, 0, stream>>>(pooled, (size_t)G * HID);
    pool_kernel<<<2048, 256, 0, stream>>>(h, graph_idx, pooled, N);
    readout_kernel<<<G, 128, 0, stream>>>(pooled, W_fc, b_fc, W_out, b_out, (float*)d_out);
}
